// MultiHeadAttention_82652350644728
// MI455X (gfx1250) — compile-verified, matrix operand order checked by reading, not run on hardware
//
#include <hip/hip_runtime.h>

// ---------------------------------------------------------------------------
// MI455X (gfx1250) multi-head causal attention, bf16 WMMA + flash softmax.
// Shapes fixed per reference: B=4, T=2048, H=16, DH=64, C=1024.
//
// Data path: weights staged to LDS by the Tensor Data Mover (double-buffered,
// TENSORcnt-tracked), activations streamed from L2 with global_load_b128,
// all matmuls on v_wmma_f32_16x16x32_bf16 (fp32 accumulate).
// ---------------------------------------------------------------------------

typedef __bf16 bf16_t;
typedef __attribute__((ext_vector_type(16))) __bf16        v16bf;
typedef __attribute__((ext_vector_type(8)))  float         v8f;
typedef __attribute__((ext_vector_type(4)))  unsigned int  u32x4;
typedef __attribute__((ext_vector_type(4)))  unsigned int  v4u;
typedef __attribute__((ext_vector_type(8)))  int           v8i;
typedef __attribute__((ext_vector_type(4)))  int           v4i;

union Frag { v16bf v; u32x4 q[2]; };

constexpr int NB = 4;      // batch
constexpr int NT = 2048;   // seq len
constexpr int NH = 16;     // heads
constexpr int ND = 64;     // head dim
constexpr int NC = 1024;   // embed dim

#if __has_builtin(__builtin_amdgcn_tensor_load_to_lds)
#define HAVE_TDM 1
#else
#define HAVE_TDM 0
#endif

__device__ __forceinline__ void wait_tensorcnt0() {
#if __has_builtin(__builtin_amdgcn_s_wait_tensorcnt)
    __builtin_amdgcn_s_wait_tensorcnt(0);
#else
    asm volatile("s_wait_tensorcnt 0x0" ::: "memory");
#endif
}

// Tell the compiler the TDM may have written this LDS object: without this,
// LLVM sees no stores to the stage buffer, folds its loads to undef, and
// deletes the ds_load_b128 B-operand reads (observed in round-3 asm).
__device__ __forceinline__ void lds_written(const void* p) {
    asm volatile("" : : "s"((unsigned long long)(size_t)p) : "memory");
}

// Issue one TDM 2-D tile load: kelems (contiguous, 2B elems) x rows, row
// stride = stride_elems, into LDS byte address lds_byte. D# per ISA §8.3/8.4.
// This toolchain exposes the 6-arg builtin:
//   (uint32x4 g0, int32x8 g1, int32x4 g2, int32x4 g3, int32x8 extra, i32 cpol)
__device__ __forceinline__ void tdm_load_2d(const void* gptr, unsigned lds_byte,
                                            unsigned kelems, unsigned rows,
                                            unsigned stride_elems) {
#if HAVE_TDM
    unsigned long long ga = (unsigned long long)(size_t)gptr;
    v4u g0;
    g0[0] = 1u;                                           // count=1, user mode
    g0[1] = lds_byte;                                     // lds_addr (bytes)
    g0[2] = (unsigned)(ga & 0xFFFFFFFFull);               // global_addr[31:0]
    g0[3] = (unsigned)((ga >> 32) & 0x1FFFFFFull)         // global_addr[56:32]
          | (2u << 30);                                   // type = 2 ("image")
    const unsigned td0 = 0x40000000u, td1 = 0x40000000u;  // huge dims: no OOB
    v8i g1;
    g1[0] = 0x00010000;                                   // data_size=1 (2B)
    g1[1] = (int)((td0 & 0xFFFFu) << 16);                 // tensor_dim0 lo
    g1[2] = (int)((td0 >> 16) | ((td1 & 0xFFFFu) << 16)); // dim0 hi | dim1 lo
    g1[3] = (int)((td1 >> 16) | (kelems << 16));          // dim1 hi | tile_dim0
    g1[4] = (int)rows;                                    // tile_dim1, tile_dim2=0
    g1[5] = (int)stride_elems;                            // tensor_dim0_stride lo
    g1[6] = 0;                                            // stride hi | dim1_stride lo
    g1[7] = 0;                                            // dim1_stride hi
    v4i z4 = {0, 0, 0, 0};
    v8i z8 = {0, 0, 0, 0, 0, 0, 0, 0};
    __builtin_amdgcn_tensor_load_to_lds(g0, g1, z4, z4, z8, 0);
#else
    (void)gptr; (void)lds_byte; (void)kelems; (void)rows; (void)stride_elems;
#endif
}

__device__ __forceinline__ bf16_t f2bf(float f) {
    unsigned u = __builtin_bit_cast(unsigned, f);
    unsigned r = u + 0x7FFFu + ((u >> 16) & 1u);          // round-to-nearest-even
    unsigned short h = (unsigned short)(r >> 16);
    return __builtin_bit_cast(bf16_t, h);
}

// Load one 16x32 bf16 A/B-operand fragment for this lane (global or LDS ptr).
// Lane l holds row (l&15); khalf=(l>>4) selects K-runs [8*khalf,8*khalf+8)
// and [16+8*khalf, 24+8*khalf) -> two contiguous 16-byte loads.
__device__ __forceinline__ v16bf load_frag(const bf16_t* rowptr, int k0, int khalf) {
    Frag f;
    f.q[0] = *(const u32x4*)(rowptr + k0 + khalf * 8);
    f.q[1] = *(const u32x4*)(rowptr + k0 + khalf * 8 + 16);
    return f.v;
}

__device__ __forceinline__ v8f wmma_bf16(v16bf a, v16bf b, v8f c) {
    return __builtin_amdgcn_wmma_f32_16x16x32_bf16(false, a, false, b,
                                                   (short)0, c, false, false);
}

// 4 WMMAs: A frag x [64 x 32] B tile staged in LDS (rows n, contiguous K=32).
__device__ __forceinline__ void btile_mma(const bf16_t* bbuf, v16bf af, v8f* acc,
                                          int lane16, int khalf) {
    #pragma unroll
    for (int nb = 0; nb < 4; ++nb) {
        Frag f;
        const bf16_t* pr = bbuf + (size_t)(nb * 16 + lane16) * 32;
        f.q[0] = *(const u32x4*)(pr + khalf * 8);
        f.q[1] = *(const u32x4*)(pr + khalf * 8 + 16);
        acc[nb] = wmma_bf16(af, f.v, acc[nb]);
    }
}

// ---------------------------------------------------------------------------
// Kernel 1: fp32 -> bf16 elementwise
// ---------------------------------------------------------------------------
__global__ void cvt_f32_bf16(const float* __restrict__ src,
                             bf16_t* __restrict__ dst, int n) {
    int i = blockIdx.x * blockDim.x + threadIdx.x;
    if (i < n) dst[i] = f2bf(src[i]);
}

// ---------------------------------------------------------------------------
// Kernel 2: fp32 [batch][R][Cc] -> bf16 [batch][Cc][R]  (weight transpose)
// ---------------------------------------------------------------------------
__global__ void cvt_transpose(const float* __restrict__ src,
                              bf16_t* __restrict__ dst, int R, int Cc) {
    int i = blockIdx.x * blockDim.x + threadIdx.x;
    int total = R * Cc;
    if (i >= total) return;
    int b = blockIdx.y;
    int r = i / Cc, c = i % Cc;
    dst[(size_t)b * total + (size_t)c * R + r] =
        f2bf(src[(size_t)b * total + (size_t)r * Cc + c]);
}

// ---------------------------------------------------------------------------
// Kernel 3: QKV projection. out[bh] (T x DH) = x[b] (T x C) * W[h] (C x DH).
// WT pre-transposed [H][DH][C]. Weight tile (64 x 32K, 4KB) staged to LDS by
// the TDM, double-buffered; waves consume via ds_load_b128 (8x reuse).
// A fragment software-pipelined one K-chunk ahead.
// transOut=1 stores output transposed [DH][T] (used for V).
// ---------------------------------------------------------------------------
__global__ __launch_bounds__(256)
void qkv_gemm(const bf16_t* __restrict__ Xbf, const bf16_t* __restrict__ WT,
              bf16_t* __restrict__ out, int transOut) {
    // Sole __shared__ object in this kernel -> LDS byte offset 0 (buf1 @ 4096).
    __shared__ bf16_t bstage[2][64 * 32];

    const int w = threadIdx.x >> 5, l = threadIdx.x & 31;
    const int lane16 = l & 15, khalf = l >> 4;
    const int bh = blockIdx.z;
    const int b = bh >> 4, h = bh & 15;                 // NH == 16
    const int m0 = blockIdx.x * 128 + w * 16;

    const bf16_t* arow = Xbf + (size_t)b * NT * NC + (size_t)(m0 + lane16) * NC;
    const bf16_t* Wh   = WT + (size_t)h * ND * NC;      // [64][1024]

    v8f acc[4] = {};
    const int NKC = NC / 32;

#if HAVE_TDM
    if (threadIdx.x < 32) {
        tdm_load_2d(Wh, 0u, 32u, 64u, NC);
        wait_tensorcnt0();
    }
    __syncthreads();
    lds_written(&bstage[0][0]);
    v16bf af = load_frag(arow, 0, khalf);               // pipelined A frag
    for (int kc = 0; kc < NKC; ++kc) {
        const int cur = kc & 1;
        if (kc + 1 < NKC && threadIdx.x < 32)
            tdm_load_2d(Wh + (kc + 1) * 32, cur ? 0u : 4096u, 32u, 64u, NC);
        if (kc + 2 < NKC) __builtin_prefetch(arow + (kc + 2) * 32, 0, 3);
        v16bf afn = af;
        if (kc + 1 < NKC) afn = load_frag(arow, (kc + 1) * 32, khalf);
        btile_mma(&bstage[cur][0], af, acc, lane16, khalf);
        if (kc + 1 < NKC) {
            if (threadIdx.x < 32) wait_tensorcnt0();
            __syncthreads();                            // next buffer ready,
            lds_written(&bstage[0][0]);                  // cur buffer released
        }
        af = afn;
    }
#else
    for (int kc = 0; kc < NKC; ++kc) {
        __syncthreads();
        {   // manual stage: 256 threads x 16B = 4KB
            int row = threadIdx.x >> 2, seg = threadIdx.x & 3;
            *(u32x4*)(&bstage[0][0] + row * 32 + seg * 8) =
                *(const u32x4*)(Wh + (size_t)row * NC + kc * 32 + seg * 8);
        }
        __syncthreads();
        v16bf af = load_frag(arow, kc * 32, khalf);
        btile_mma(&bstage[0][0], af, acc, lane16, khalf);
    }
#endif

    const size_t obase = (size_t)bh * NT * ND;
    #pragma unroll
    for (int nb = 0; nb < 4; ++nb)
        #pragma unroll
        for (int r = 0; r < 8; ++r) {
            int t = m0 + r + 8 * khalf;                 // C-layout row
            int d = nb * 16 + lane16;                   // C-layout col
            bf16_t vb = f2bf(acc[nb][r]);
            if (transOut) out[obase + (size_t)d * NT + t] = vb;
            else          out[obase + (size_t)t * ND + d] = vb;
        }
}

// ---------------------------------------------------------------------------
// Kernel 4: flash attention per (b,h). One wave owns 16 query rows; streams
// 32-wide K/V tiles with causal mask + online softmax (exp2 domain).
// Q,K: [BH][T][DH] bf16.  VT: [BH][DH][T] bf16.  O: [B][T][H*DH] bf16.
// ---------------------------------------------------------------------------
__global__ __launch_bounds__(256)
void flash_attn(const bf16_t* __restrict__ Q, const bf16_t* __restrict__ K,
                const bf16_t* __restrict__ VT, bf16_t* __restrict__ Obf) {
    __shared__ bf16_t lds_p[8][16][32];                 // 8 KB, 1 KB per wave

    const int w = threadIdx.x >> 5, l = threadIdx.x & 31;
    const int lane16 = l & 15, khalf = l >> 4;
    const int bh = blockIdx.y;
    const int trow0 = blockIdx.x * 128 + w * 16;

    const bf16_t* Qb = Q  + (size_t)bh * NT * ND;
    const bf16_t* Kb = K  + (size_t)bh * NT * ND;
    const bf16_t* Vb = VT + (size_t)bh * ND * NT;

    // Q fragments (DH=64 -> two K-chunks of 32), live for the whole loop.
    const bf16_t* qrow = Qb + (size_t)(trow0 + lane16) * ND;
    v16bf qf0 = load_frag(qrow, 0, khalf);
    v16bf qf1 = load_frag(qrow, 32, khalf);

    v8f acc_o[4] = {};
    float mstat[8], lstat[8];
    #pragma unroll
    for (int r = 0; r < 8; ++r) { mstat[r] = -__builtin_inff(); lstat[r] = 0.f; }

    const float sc = 0.125f * 1.44269504088896340736f;  // (1/sqrt(DH))*log2(e)
    const int ntiles = (trow0 >> 5) + 1;                // causal bound

    for (int j = 0; j < ntiles; ++j) {
        const int s0 = j * 32;

        // Prefetch next tile's K/V rows while this tile computes.
        if (j + 1 < ntiles) {
            __builtin_prefetch(Kb + (size_t)(s0 + 32 + lane16) * ND, 0, 3);
            __builtin_prefetch(Vb + (size_t)(l * 2) * NT + s0 + 32, 0, 3);
        }

        // S tile (16 x 32) = Q (16 x 64) * K^T : two N-blocks, two K-chunks.
        v8f sa[2] = {};
        #pragma unroll
        for (int nb = 0; nb < 2; ++nb) {
            const bf16_t* krow = Kb + (size_t)(s0 + nb * 16 + lane16) * ND;
            sa[nb] = wmma_bf16(qf0, load_frag(krow, 0,  khalf), sa[nb]);
            sa[nb] = wmma_bf16(qf1, load_frag(krow, 32, khalf), sa[nb]);
        }

        // Scale into log2 domain, causal mask, online max.
        float mnew[8];
        #pragma unroll
        for (int r = 0; r < 8; ++r) {
            int t = trow0 + r + 8 * khalf;
            float v0 = sa[0][r] * sc;
            float v1 = sa[1][r] * sc;
            if (s0 + lane16      > t) v0 = -__builtin_inff();
            if (s0 + 16 + lane16 > t) v1 = -__builtin_inff();
            sa[0][r] = v0; sa[1][r] = v1;
            float tm = fmaxf(v0, v1);
            #pragma unroll
            for (int mk = 1; mk < 16; mk <<= 1)
                tm = fmaxf(tm, __shfl_xor(tm, mk, 32));  // within 16-lane group
            mnew[r] = fmaxf(mstat[r], tm);
        }

        // p = exp2(s - mnew); rescale running sum & output accumulator.
        #pragma unroll
        for (int r = 0; r < 8; ++r) {
            float alpha = exp2f(mstat[r] - mnew[r]);
            float p0 = exp2f(sa[0][r] - mnew[r]);
            float p1 = exp2f(sa[1][r] - mnew[r]);
            sa[0][r] = p0; sa[1][r] = p1;
            float rs = p0 + p1;
            #pragma unroll
            for (int mk = 1; mk < 16; mk <<= 1)
                rs += __shfl_xor(rs, mk, 32);
            lstat[r] = lstat[r] * alpha + rs;
            mstat[r] = mnew[r];
            #pragma unroll
            for (int nb2 = 0; nb2 < 4; ++nb2) acc_o[nb2][r] *= alpha;
        }

        // C-layout -> A-layout for P via per-wave LDS round trip.
        #pragma unroll
        for (int nb = 0; nb < 2; ++nb)
            #pragma unroll
            for (int r = 0; r < 8; ++r)
                lds_p[w][r + 8 * khalf][nb * 16 + lane16] = f2bf(sa[nb][r]);

        // Per-wave DS ops are in program order; block compiler reordering.
        asm volatile("" ::: "memory");

        v16bf pf;
        {
            Frag f;
            const bf16_t* pr = &lds_p[w][lane16][0];
            f.q[0] = *(const u32x4*)(pr + khalf * 8);
            f.q[1] = *(const u32x4*)(pr + khalf * 8 + 16);
            pf = f.v;
        }

        // O (16 x 64) += P (16 x 32) * V (32 x 64); B operand rows = VT rows.
        #pragma unroll
        for (int nb2 = 0; nb2 < 4; ++nb2) {
            const bf16_t* vrow = Vb + (size_t)(nb2 * 16 + lane16) * NT;
            Frag f;
            f.q[0] = *(const u32x4*)(vrow + s0 + khalf * 8);
            f.q[1] = *(const u32x4*)(vrow + s0 + khalf * 8 + 16);
            acc_o[nb2] = wmma_bf16(pf, f.v, acc_o[nb2]);
        }
    }

    // Normalize and store concat-head layout [B][T][H*DH].
    const int b = bh >> 4, h = bh & 15;
    #pragma unroll
    for (int nb2 = 0; nb2 < 4; ++nb2)
        #pragma unroll
        for (int r = 0; r < 8; ++r) {
            int t = trow0 + r + 8 * khalf;
            int d = nb2 * 16 + lane16;
            float o = acc_o[nb2][r] / lstat[r];
            Obf[((size_t)b * NT + t) * NC + h * ND + d] = f2bf(o);
        }
}

// ---------------------------------------------------------------------------
// Kernel 5: output projection. out (M x C, fp32) = A (M x C, bf16) * Wo + bo.
// WoT pre-transposed [C][C]; same TDM-staged weight-tile scheme as qkv_gemm.
// ---------------------------------------------------------------------------
__global__ __launch_bounds__(256)
void oproj_gemm(const bf16_t* __restrict__ Abf, const bf16_t* __restrict__ WoT,
                const float* __restrict__ bo, float* __restrict__ out) {
    __shared__ bf16_t bstage[2][64 * 32];               // LDS offset 0 assumed

    const int w = threadIdx.x >> 5, l = threadIdx.x & 31;
    const int lane16 = l & 15, khalf = l >> 4;
    const int m0 = blockIdx.x * 128 + w * 16;
    const int n0 = blockIdx.y * 64;

    const bf16_t* arow = Abf + (size_t)(m0 + lane16) * NC;
    const bf16_t* Wn   = WoT + (size_t)n0 * NC;

    v8f acc[4] = {};
    const int NKC = NC / 32;

#if HAVE_TDM
    if (threadIdx.x < 32) {
        tdm_load_2d(Wn, 0u, 32u, 64u, NC);
        wait_tensorcnt0();
    }
    __syncthreads();
    lds_written(&bstage[0][0]);
    v16bf af = load_frag(arow, 0, khalf);               // pipelined A frag
    for (int kc = 0; kc < NKC; ++kc) {
        const int cur = kc & 1;
        if (kc + 1 < NKC && threadIdx.x < 32)
            tdm_load_2d(Wn + (kc + 1) * 32, cur ? 0u : 4096u, 32u, 64u, NC);
        if (kc + 2 < NKC) __builtin_prefetch(arow + (kc + 2) * 32, 0, 3);
        v16bf afn = af;
        if (kc + 1 < NKC) afn = load_frag(arow, (kc + 1) * 32, khalf);
        btile_mma(&bstage[cur][0], af, acc, lane16, khalf);
        if (kc + 1 < NKC) {
            if (threadIdx.x < 32) wait_tensorcnt0();
            __syncthreads();
            lds_written(&bstage[0][0]);
        }
        af = afn;
    }
#else
    for (int kc = 0; kc < NKC; ++kc) {
        __syncthreads();
        {
            int row = threadIdx.x >> 2, seg = threadIdx.x & 3;
            *(u32x4*)(&bstage[0][0] + row * 32 + seg * 8) =
                *(const u32x4*)(Wn + (size_t)row * NC + kc * 32 + seg * 8);
        }
        __syncthreads();
        v16bf af = load_frag(arow, kc * 32, khalf);
        btile_mma(&bstage[0][0], af, acc, lane16, khalf);
    }
#endif

    #pragma unroll
    for (int nb = 0; nb < 4; ++nb) {
        int n = n0 + nb * 16 + lane16;
        float bias = bo[n];
        #pragma unroll
        for (int r = 0; r < 8; ++r) {
            int m = m0 + r + 8 * khalf;
            out[(size_t)m * NC + n] = acc[nb][r] + bias;
        }
    }
}

// ---------------------------------------------------------------------------
// Host launcher
// ---------------------------------------------------------------------------
extern "C" void kernel_launch(void* const* d_in, const int* in_sizes, int n_in,
                              void* d_out, int out_size, void* d_ws, size_t ws_size,
                              hipStream_t stream) {
    const float* x  = (const float*)d_in[0];
    const float* Wq = (const float*)d_in[1];
    const float* Wk = (const float*)d_in[2];
    const float* Wv = (const float*)d_in[3];
    const float* Wo = (const float*)d_in[4];
    const float* bo = (const float*)d_in[5];
    float* out = (float*)d_out;

    // Workspace carve-up (bf16 elements); total ~92 MB.
    bf16_t* ws = (bf16_t*)d_ws;
    size_t off = 0;
    bf16_t* x_bf  = ws + off; off += (size_t)NB * NT * NC;
    bf16_t* WqT   = ws + off; off += (size_t)NH * ND * NC;
    bf16_t* WkT   = ws + off; off += (size_t)NH * ND * NC;
    bf16_t* WvT   = ws + off; off += (size_t)NH * ND * NC;
    bf16_t* WoT   = ws + off; off += (size_t)NC * NC;
    bf16_t* q_bf  = ws + off; off += (size_t)NB * NH * NT * ND;
    bf16_t* k_bf  = ws + off; off += (size_t)NB * NH * NT * ND;
    bf16_t* vT_bf = ws + off; off += (size_t)NB * NH * NT * ND;
    bf16_t* a_bf  = ws + off; off += (size_t)NB * NT * NC;
    (void)in_sizes; (void)n_in; (void)out_size; (void)ws_size;

    // 1) conversions
    {
        int n = NB * NT * NC;
        cvt_f32_bf16<<<(n + 255) / 256, 256, 0, stream>>>(x, x_bf, n);
    }
    {
        dim3 g((NC * ND + 255) / 256, NH);
        cvt_transpose<<<g, 256, 0, stream>>>(Wq, WqT, NC, ND);
        cvt_transpose<<<g, 256, 0, stream>>>(Wk, WkT, NC, ND);
        cvt_transpose<<<g, 256, 0, stream>>>(Wv, WvT, NC, ND);
    }
    {
        dim3 g((NC * NC + 255) / 256, 1);
        cvt_transpose<<<g, 256, 0, stream>>>(Wo, WoT, NC, NC);
    }

    // 2) QKV projections (V stored transposed [DH][T] for the p*v WMMA).
    {
        dim3 g(NT / 128, 1, NB * NH);
        qkv_gemm<<<g, 256, 0, stream>>>(x_bf, WqT, q_bf, 0);
        qkv_gemm<<<g, 256, 0, stream>>>(x_bf, WkT, k_bf, 0);
        qkv_gemm<<<g, 256, 0, stream>>>(x_bf, WvT, vT_bf, 1);
    }

    // 3) flash attention
    {
        dim3 g(NT / 128, NB * NH);
        flash_attn<<<g, 256, 0, stream>>>(q_bf, k_bf, vT_bf, a_bf);
    }

    // 4) output projection + bias (fp32 result)
    {
        dim3 g((NB * NT) / 128, NC / 64);
        oproj_gemm<<<g, 256, 0, stream>>>(a_bf, WoT, bo, out);
    }
}
